// GAttACNN_53249004536071
// MI455X (gfx1250) — compile-verified, matrix operand order checked by reading, not run on hardware
//
#include <hip/hip_runtime.h>

typedef __attribute__((ext_vector_type(16))) __bf16 v16bf;
typedef __attribute__((ext_vector_type(8)))  __bf16 v8bf;
typedef __attribute__((ext_vector_type(8)))  float  v8f;

#define EPSV 1e-6f

__device__ __forceinline__ __bf16 tobf(float x) { return (__bf16)x; }

__device__ __forceinline__ v8f wmma_bf16(v16bf a, v16bf b, v8f c) {
  return __builtin_amdgcn_wmma_f32_16x16x32_bf16(false, a, false, b, (short)0, c,
                                                 false, false);
}

__device__ __forceinline__ v16bf cat16(v8bf lo, v8bf hi) {
  return __builtin_shufflevector(lo, hi, 0, 1, 2, 3, 4, 5, 6, 7,
                                 8, 9, 10, 11, 12, 13, 14, 15);
}

// ================= weight prep: bf16, WMMA-friendly layouts =================
// w2t[m*576 + tap*64 + c] = bf16(w2[m*576 + c*9 + tap])   (tap-major K)
__global__ void k_prep_w2t(const float* __restrict__ w2, __bf16* __restrict__ w2t) {
  const int m = blockIdx.x;
  for (int r = threadIdx.x; r < 576; r += 256) {
    int tap = r >> 6, c = r & 63;
    w2t[m * 576 + r] = tobf(w2[m * 576 + c * 9 + tap]);
  }
}

// wkt[d*C + c] = bf16(wk[c*32 + d])   (transpose to d-major rows)
__global__ void k_prep_wkt(const float* __restrict__ wk, __bf16* __restrict__ wkt,
                           int C, int lg) {
  int idx = blockIdx.x * 256 + threadIdx.x;   // 32*C elements
  int c = idx & (C - 1);
  int d = idx >> lg;
  wkt[idx] = tobf(wk[c * 32 + d]);
}

// straight f32 -> bf16 convert
__global__ void k_prep_bf(const float* __restrict__ src, __bf16* __restrict__ dst, int n) {
  int idx = blockIdx.x * 256 + threadIdx.x;
  if (idx < n) dst[idx] = tobf(src[idx]);
}

// ================= conv1: 3->64, 3x3 SAME, ReLU (direct; K=27) ==============
__global__ void k_conv1(const float* __restrict__ x, const float* __restrict__ w,
                        const float* __restrict__ bias, float* __restrict__ f1) {
  int idx = blockIdx.x * 256 + threadIdx.x;   // [0, 16*64*16384)
  int pix = idx & 16383;
  int o   = (idx >> 14) & 63;
  int b   = idx >> 20;
  int y = pix >> 7, xx = pix & 127;
  float acc = bias[o];
#pragma unroll
  for (int c = 0; c < 3; ++c) {
    const float* xp = x + ((b * 3 + c) << 14);
    const float* wp = w + (o * 3 + c) * 9;
#pragma unroll
    for (int dy = -1; dy <= 1; ++dy) {
      int ys = y + dy;
      if ((unsigned)ys < 128u) {
#pragma unroll
        for (int dx = -1; dx <= 1; ++dx) {
          int xs = xx + dx;
          if ((unsigned)xs < 128u)
            acc += xp[(ys << 7) + xs] * wp[(dy + 1) * 3 + (dx + 1)];
        }
      }
    }
  }
  f1[idx] = fmaxf(acc, 0.f);
}

// ================= 2x2 maxpool -> bf16 ======================================
__global__ void k_pool(const float* __restrict__ f1, __bf16* __restrict__ p1) {
  int idx = blockIdx.x * 256 + threadIdx.x;   // 16*64*4096
  int po = idx & 4095;
  int c  = (idx >> 12) & 63;
  int b  = idx >> 18;
  int yo = po >> 6, xo = po & 63;
  const float* s = f1 + (((b * 64 + c) << 14) + (yo << 8) + (xo << 1));
  p1[idx] = tobf(fmaxf(fmaxf(s[0], s[1]), fmaxf(s[128], s[129])));
}

// ===== conv2 implicit GEMM via WMMA: M=128, N=16 px/block, K=576 (tap-major)
__global__ void k_conv2_wmma(const __bf16* __restrict__ p1, const __bf16* __restrict__ w2t,
                             const float* __restrict__ b2, float* __restrict__ f2) {
  __shared__ __attribute__((aligned(16))) __bf16 Bs[16 * 32];   // [n][k]
  const int tid  = threadIdx.x;
  const int lane = tid & 31, wave = tid >> 5;
  const int ng0  = blockIdx.x << 4;     // 16 contiguous pixels, same batch & row
  const int b    = ng0 >> 12;
  const int pix0 = ng0 & 4095;
  const int y0     = pix0 >> 6;
  const int x0base = pix0 & 63;
  const int mrow = (wave << 4) + (lane & 15);
  const int kb   = (lane >> 4) << 3;    // 0 or 8 (per-lane-half K sub-offset)
  const int ln   = lane & 15;
  const int sn   = tid & 15;            // staging: pixel column
  const int sk   = tid >> 4;            // staging: k row (handles sk, sk+16)
  v8f acc = {};
  for (int ks = 0; ks < 18; ++ks) {
    const int k0 = ks << 5;
    __syncthreads();
    // stage 32x16 bf16 B tile (im2col of p1, tap-major K) into LDS, [n][k]
#pragma unroll
    for (int h = 0; h < 2; ++h) {
      int kk  = sk + (h << 4);
      int K   = k0 + kk;
      int tap = K >> 6;                 // shifts only: no integer division
      int c   = K & 63;
      int dy  = (tap >= 6) ? 1 : ((tap >= 3) ? 0 : -1);
      int dx  = tap - (dy + 1) * 3 - 1;
      int ys = y0 + dy, xs = x0base + sn + dx;
      __bf16 v = (__bf16)0.f;
      if (((unsigned)ys < 64u) & ((unsigned)xs < 64u))
        v = p1[((b * 64 + c) * 64 + ys) * 64 + xs];
      Bs[sn * 32 + kk] = v;
    }
    if (ks < 17) __builtin_prefetch(w2t + mrow * 576 + k0 + 32, 0, 0);
    __syncthreads();
    // A fragment: contiguous bf16, two 16B vector loads
    const __bf16* ap = w2t + mrow * 576 + k0 + kb;
    v16bf a   = cat16(*(const v8bf*)ap, *(const v8bf*)(ap + 16));
    // B fragment: contiguous per-lane column from LDS, two 16B ds loads
    const __bf16* bp = &Bs[ln * 32 + kb];
    v16bf bfr = cat16(*(const v8bf*)bp, *(const v8bf*)(bp + 16));
    acc = wmma_bf16(a, bfr, acc);
  }
  // epilogue: bias + ReLU; C/D layout: VGPR j -> row j (+8 for upper lanes)
  const int half = (lane >> 4) << 3;
  const int pixg = pix0 + ln;
#pragma unroll
  for (int j = 0; j < 8; ++j) {
    int m = (wave << 4) + j + half;
    float v = fmaxf(acc[j] + b2[m], 0.f);
    f2[((b * 128 + m) << 12) + pixg] = v;
  }
}

// ================= per-(b,c) spatial sum ====================================
__global__ void k_rowsum(const float* __restrict__ src, float* __restrict__ dst, int n) {
  __shared__ float sm[256];
  const float* p = src + (size_t)blockIdx.x * n;
  float s = 0.f;
  for (int i = threadIdx.x; i < n; i += 256) s += p[i];
  sm[threadIdx.x] = s;
  __syncthreads();
  for (int off = 128; off > 0; off >>= 1) {
    if (threadIdx.x < off) sm[threadIdx.x] += sm[threadIdx.x + off];
    __syncthreads();
  }
  if (threadIdx.x == 0) dst[blockIdx.x] = sm[0];
}

// ====== head: fd GEMM (WMMA) + q1/q2/qd/kd + qg + fdm (one block) ===========
__global__ void k_head(const float* __restrict__ s1, const float* __restrict__ s2,
                       const float* __restrict__ fw1, const float* __restrict__ fb1,
                       const float* __restrict__ wq1, const float* __restrict__ wq2,
                       const float* __restrict__ wkd, const float* __restrict__ wqd,
                       const float* __restrict__ gd, float* __restrict__ fd_ws,
                       __bf16* __restrict__ fd_bf, float* __restrict__ qg_ws,
                       float* __restrict__ fdm_out) {
  __shared__ float qgs[512];
  __shared__ float kds[512];
  __shared__ float nrm[16];
  __shared__ float agr[16];
  const int tid = threadIdx.x, lane = tid & 31, wave = tid >> 5;
  const int ln = lane & 15;
  const int kb = (lane >> 4) << 3;
  // fd = relu((s2/4096) @ fw1 + fb1) : M=16, N=1024, K=128
  for (int it = 0; it < 8; ++it) {
    const int n0 = (wave + (it << 3)) << 4;
    v8f acc = {};
    for (int ks = 0; ks < 4; ++ks) {
      const int k0 = ks << 5;
      v16bf a, bb;
#pragma unroll
      for (int i = 0; i < 8; ++i) {
        a[i]      = tobf(s2[ln * 128 + k0 + kb + i] * (1.f / 4096.f));
        a[8 + i]  = tobf(s2[ln * 128 + k0 + kb + 16 + i] * (1.f / 4096.f));
        bb[i]     = tobf(fw1[(k0 + kb + i) * 1024 + n0 + ln]);
        bb[8 + i] = tobf(fw1[(k0 + kb + 16 + i) * 1024 + n0 + ln]);
      }
      acc = wmma_bf16(a, bb, acc);
    }
    const int col  = n0 + ln;
    const int half = (lane >> 4) << 3;
#pragma unroll
    for (int j = 0; j < 8; ++j) {
      int br = j + half;  // batch row
      float v = fmaxf(acc[j] + fb1[col], 0.f);
      fd_ws[br * 1024 + col] = v;
      fd_bf[br * 1024 + col] = tobf(v);
    }
  }
  __threadfence();
  __syncthreads();
  // q1,q2,qd,kd (tiny GEMVs)
  for (int t = tid; t < 512; t += 256) {
    int b = t >> 5, d = t & 31;
    float q1v = 0.f, q2v = 0.f, qdv = 0.f, kdv = 0.f;
    for (int c = 0; c < 64; ++c)  q1v += s1[b * 64 + c] * wq1[c * 32 + d];
    q1v *= (1.f / 16384.f);
    for (int c = 0; c < 128; ++c) q2v += s2[b * 128 + c] * wq2[c * 32 + d];
    q2v *= (1.f / 4096.f);
    for (int k = 0; k < 1024; ++k) {
      float f = fd_ws[b * 1024 + k];
      qdv += f * wqd[k * 32 + d];
      kdv += f * wkd[k * 32 + d];
    }
    qgs[t] = (q1v + q2v + qdv) * (1.f / 3.f);
    kds[t] = kdv;
  }
  __syncthreads();
  if (tid < 16) {
    float s = 0.f;
    for (int d = 0; d < 32; ++d) { float v = qgs[tid * 32 + d]; s += v * v; }
    nrm[tid] = sqrtf(s) + EPSV;
  }
  __syncthreads();
  for (int t = tid; t < 512; t += 256) {
    float v = qgs[t] / nrm[t >> 5];
    qgs[t]   = v;
    qg_ws[t] = v;
  }
  __syncthreads();
  if (tid < 16) {
    float ss = 0.f, dot = 0.f;
    for (int d = 0; d < 32; ++d) {
      float k = kds[tid * 32 + d];
      ss  += k * k;
      dot += k * qgs[tid * 32 + d];
    }
    agr[tid] = dot / (sqrtf(ss) + EPSV);
  }
  __syncthreads();
  const float gdv = *gd;
  for (int i = tid; i < 16384; i += 256) {
    int b = i >> 10;
    fdm_out[i] = fd_ws[i] * (1.f + gdv * agr[b]);
  }
}

// ====== logits = fd @ fw2 + fb2 : M=16, N=1000, K=1024 (WMMA, bf16 inputs) ==
__global__ void k_fc2(const __bf16* __restrict__ fd_bf, const __bf16* __restrict__ fw2bf,
                      const float* __restrict__ fb2, float* __restrict__ logits) {
  const int lane = threadIdx.x & 31, wave = threadIdx.x >> 5;
  const int ntile = blockIdx.x * 8 + wave;
  if (ntile >= 63) return;              // wave-uniform exit
  const int n0 = ntile << 4;
  const int ln = lane & 15;
  const int kb = (lane >> 4) << 3;
  const int colc = min(n0 + ln, 999);   // clamp loads for the ragged tile
  v8f acc = {};
  for (int ks = 0; ks < 32; ++ks) {
    const int k0 = ks << 5;
    const __bf16* ap = fd_bf + ln * 1024 + k0 + kb;
    v16bf a = cat16(*(const v8bf*)ap, *(const v8bf*)(ap + 16));
    v16bf bb;
#pragma unroll
    for (int i = 0; i < 8; ++i) {
      bb[i]     = fw2bf[(k0 + kb + i) * 1000 + colc];
      bb[8 + i] = fw2bf[(k0 + kb + 16 + i) * 1000 + colc];
    }
    acc = wmma_bf16(a, bb, acc);
  }
  const int half = (lane >> 4) << 3;
  const int col  = n0 + ln;
  if (col < 1000) {
#pragma unroll
    for (int j = 0; j < 8; ++j) {
      int b = j + half;
      logits[b * 1000 + col] = acc[j] + fb2[col];
    }
  }
}

// ====== key projection (WMMA, D=32) + agreement + in-place modulation =======
__global__ void k_modulate(float* __restrict__ f, const __bf16* __restrict__ wkt,
                           const float* __restrict__ qg, const float* __restrict__ gptr,
                           int C, int HW) {
  const int lane = threadIdx.x & 31, wave = threadIdx.x >> 5;
  const int ng0  = (blockIdx.x * 8 + wave) << 4;  // 16 contiguous pixels
  const int b    = ng0 / HW;
  const int pix0 = ng0 - b * HW;
  const int ln = lane & 15;
  const int kb = (lane >> 4) << 3;
  v8f acc0 = {}, acc1 = {};
  const int ksteps = C >> 5;
  for (int ks = 0; ks < ksteps; ++ks) {
    const int k0 = ks << 5;
    // B: activation column per pixel (strided; coalesced across lanes)
    v16bf bfr;
#pragma unroll
    for (int i = 0; i < 8; ++i) {
      bfr[i]     = tobf(f[(b * C + k0 + kb + i) * HW + pix0 + ln]);
      bfr[8 + i] = tobf(f[(b * C + k0 + kb + 16 + i) * HW + pix0 + ln]);
    }
    // A: pre-transposed bf16 wk rows (contiguous 16B vector loads)
    const __bf16* a0p = wkt + ln * C + k0 + kb;
    const __bf16* a1p = wkt + (16 + ln) * C + k0 + kb;
    v16bf a0 = cat16(*(const v8bf*)a0p, *(const v8bf*)(a0p + 16));
    v16bf a1 = cat16(*(const v8bf*)a1p, *(const v8bf*)(a1p + 16));
    acc0 = wmma_bf16(a0, bfr, acc0);   // d = 0..15
    acc1 = wmma_bf16(a1, bfr, acc1);   // d = 16..31
  }
  const int half = (lane >> 4) << 3;
  float ss = 0.f, dot = 0.f;
#pragma unroll
  for (int j = 0; j < 8; ++j) {
    float v0 = acc0[j];
    float v1 = acc1[j];
    ss  += v0 * v0 + v1 * v1;
    dot += v0 * qg[b * 32 + j + half] + v1 * qg[b * 32 + 16 + j + half];
  }
  ss  += __shfl_xor(ss, 16);
  dot += __shfl_xor(dot, 16);
  const float scale = 1.f + (*gptr) * (dot / (sqrtf(ss) + EPSV));
  const int chalf = lane >> 4;
  for (int cc = 0; cc < (C >> 1); ++cc) {
    int c = (cc << 1) + chalf;
    int idx = (b * C + c) * HW + pix0 + ln;
    f[idx] *= scale;
  }
}

extern "C" void kernel_launch(void* const* d_in, const int* in_sizes, int n_in,
                              void* d_out, int out_size, void* d_ws, size_t ws_size,
                              hipStream_t stream) {
  const float* x   = (const float*)d_in[0];
  const float* cw1 = (const float*)d_in[1];
  const float* cb1 = (const float*)d_in[2];
  const float* cw2 = (const float*)d_in[3];
  const float* cb2 = (const float*)d_in[4];
  const float* fw1 = (const float*)d_in[5];
  const float* fb1 = (const float*)d_in[6];
  const float* fw2 = (const float*)d_in[7];
  const float* fb2 = (const float*)d_in[8];
  const float* wk1 = (const float*)d_in[9];
  const float* wq1 = (const float*)d_in[10];
  const float* wk2 = (const float*)d_in[11];
  const float* wq2 = (const float*)d_in[12];
  const float* wkd = (const float*)d_in[13];
  const float* wqd = (const float*)d_in[14];
  const float* g1  = (const float*)d_in[15];
  const float* g2  = (const float*)d_in[16];
  const float* gd  = (const float*)d_in[17];

  float* out    = (float*)d_out;
  float* logits = out;                                  // 16*1000
  float* f1m    = out + 16000;                          // 16*64*128*128
  float* f2m    = out + 16000 + 16777216;               // 16*128*64*64
  float* fdm    = out + 16000 + 16777216 + 8388608;     // 16*1024

  // ---- workspace: small fp32 region, then bf16 region ----
  float* wsf = (float*)d_ws;
  float* s1 = wsf;                  // 1024
  float* s2 = s1 + 1024;            // 2048
  float* fd = s2 + 2048;            // 16384
  float* qg = fd + 16384;           // 512
  __bf16* wsb  = (__bf16*)(wsf + 19968);   // 16B-aligned boundary
  __bf16* p1   = wsb;               // 4194304
  __bf16* w2t  = p1 + 4194304;      // 73728
  __bf16* wkt1 = w2t + 73728;       // 2048
  __bf16* wkt2 = wkt1 + 2048;       // 4096
  __bf16* fw2b = wkt2 + 4096;       // 1024000
  __bf16* fdbf = fw2b + 1024000;    // 16384

  // weight prep (independent, cheap)
  k_prep_w2t<<<128, 256, 0, stream>>>(cw2, w2t);
  k_prep_wkt<<<8, 256, 0, stream>>>(wk1, wkt1, 64, 6);
  k_prep_wkt<<<16, 256, 0, stream>>>(wk2, wkt2, 128, 7);
  k_prep_bf<<<4000, 256, 0, stream>>>(fw2, fw2b, 1024000);

  k_conv1<<<65536, 256, 0, stream>>>(x, cw1, cb1, f1m);
  k_pool<<<16384, 256, 0, stream>>>(f1m, p1);
  k_conv2_wmma<<<4096, 256, 0, stream>>>(p1, w2t, cb2, f2m);
  k_rowsum<<<1024, 256, 0, stream>>>(f1m, s1, 16384);
  k_rowsum<<<2048, 256, 0, stream>>>(f2m, s2, 4096);
  k_head<<<1, 256, 0, stream>>>(s1, s2, fw1, fb1, wq1, wq2, wkd, wqd, gd,
                                fd, fdbf, qg, fdm);
  k_fc2<<<8, 256, 0, stream>>>(fdbf, fw2b, fb2, logits);
  k_modulate<<<2048, 256, 0, stream>>>(f1m, wkt1, qg, g1, 64, 16384);
  k_modulate<<<512, 256, 0, stream>>>(f2m, wkt2, qg, g2, 128, 4096);
}
